// RSSITransformerModel_36498632081748
// MI455X (gfx1250) — matrix operand order repaired and checked by reading, not compiled, not run on hardware
//
#include <hip/hip_runtime.h>
#include <hip/hip_bf16.h>
#include <math.h>
#include <stdint.h>

// ---------------- problem constants (match reference) ----------------
#define NCHILD 60000
#define NAP    15000
#define NEDGE  200000
#define NG     128
#define FDIM   384      // H * HID = 6 * 64
#define NHEAD  6
#define HIDC   64
#define EMBED  32
#define APEMB  175

typedef __attribute__((ext_vector_type(16))) __bf16          bf16x16;
typedef __attribute__((ext_vector_type(8)))  float           f32x8;

__device__ __forceinline__ unsigned short f2bf(float f) {
    unsigned u = __float_as_uint(f);
    u += 0x7FFFu + ((u >> 16) & 1u);      // round-to-nearest-even
    return (unsigned short)(u >> 16);
}

// float atomic max via sign-split int atomics (dst initialized to -inf)
__device__ __forceinline__ void atomicMaxF(float* a, float v) {
    if (v >= 0.0f) atomicMax((int*)a, __float_as_int(v));
    else           atomicMin((unsigned int*)a, __float_as_uint(v));
}

// ---------------- small utility kernels ----------------
__global__ void fill_f32(float* __restrict__ p, float v, long n) {
    long t = (long)blockIdx.x * blockDim.x + threadIdx.x;
    if (t < n) p[t] = v;
}

// elementwise f32 -> bf16 (activation pre-conversion, done once per matrix)
__global__ void convert_x(const float* __restrict__ X, unsigned short* __restrict__ Xb, long n) {
    long t = (long)blockIdx.x * blockDim.x + threadIdx.x;
    if (t < n) Xb[t] = f2bf(X[t]);
}

// xa[i, 0:32] = ap_embed[i % 175, 0:32]
__global__ void gather_ap_embed(const float* __restrict__ emb, float* __restrict__ xa) {
    long t = (long)blockIdx.x * blockDim.x + threadIdx.x;
    if (t >= (long)NAP * EMBED) return;
    int i = (int)(t >> 5), c = (int)(t & 31);
    xa[t] = emb[(size_t)(i % APEMB) * EMBED + c];
}

// Y[i, c] = x[i] * W[c] + b[c]   (K == 1 linears, layer 0 child side)
__global__ void lin_k1(const float* __restrict__ x, const float* __restrict__ W,
                       const float* __restrict__ b, float* __restrict__ Y, int n) {
    long t = (long)blockIdx.x * blockDim.x + threadIdx.x;
    if (t >= (long)n * FDIM) return;
    int i = (int)(t / FDIM), c = (int)(t % FDIM);
    Y[t] = x[i] * W[c] + b[c];
}

// Wt[k * outd + n] = bf16( W[n * ind + k] )   (transpose + cvt for WMMA B)
__global__ void convert_wt(const float* __restrict__ W, unsigned short* __restrict__ Wt,
                           int outd, int ind) {
    long t = (long)blockIdx.x * blockDim.x + threadIdx.x;
    if (t >= (long)outd * ind) return;
    int k = (int)(t / outd), ncol = (int)(t % outd);
    Wt[t] = f2bf(W[(size_t)ncol * ind + k]);
}

// ---------------- WMMA GEMM: Y[n,N] = Xb[n,K](bf16) @ Wt[K,N](bf16) + bias[N] ----------------
// One wave per block; 16(M) x 64(N) strip per wave: 4 WMMA chains share one A
// fragment per K-step. Software pipelining with two STATICALLY-NAMED fragment
// sets (a0/b0, a1/b1) and a 2x-unrolled stage loop so all fragments stay in
// VGPRs (no dynamic array indexing -> no scratch spill). K/32 is 1, 2 or 12:
// pair-loop covers even step counts, steps==1 takes the direct path.
// Pointer-increment addressing. M tail: loads clamped, stores guarded.
__global__ __launch_bounds__(32)
void gemm_wmma(const unsigned short* __restrict__ Xb, const unsigned short* __restrict__ Wt,
               const float* __restrict__ bias, float* __restrict__ Y,
               int n, int K, int N) {
    const int lane  = threadIdx.x;
    const int m0    = blockIdx.x * 16;
    const int nb    = blockIdx.y * 64;
    const int mrow  = min(m0 + (lane & 15), n - 1);   // clamp: branch-free loads
    const int g8    = (lane >> 4) * 8;                // A-fragment K swizzle base
    const int col   = lane & 15;

    f32x8 c[4];
#pragma unroll
    for (int j = 0; j < 4; ++j) {
        float bv = bias[nb + 16 * j + col];
#pragma unroll
        for (int r = 0; r < 8; ++r) c[j][r] = bv;
    }

    union U { uint4 q[2]; bf16x16 bf; };
    U a0, a1, b0[4], b1[4];
    const unsigned short* xptr = Xb + (size_t)mrow * K + g8;   // +32 per stage
    const unsigned short* wptr = Wt + (size_t)g8 * N + nb + col;   // +32*N per stage
    const size_t wstep = (size_t)32 * N;

    auto ldA = [&](U& A) {
        A.q[0] = *(const uint4*)(xptr);
        A.q[1] = *(const uint4*)(xptr + 16);
        xptr += 32;
    };
    auto ldB = [&](U (&B)[4]) {
#pragma unroll
        for (int j = 0; j < 4; ++j) {
#pragma unroll
            for (int s = 0; s < 8; ++s) {
                B[j].bf[s]     = __builtin_bit_cast(__bf16, wptr[(size_t)s * N + 16 * j]);
                B[j].bf[s + 8] = __builtin_bit_cast(__bf16, wptr[(size_t)(s + 16) * N + 16 * j]);
            }
        }
        wptr += wstep;
    };
    auto mma = [&](U& A, U (&B)[4]) {
#pragma unroll
        for (int j = 0; j < 4; ++j)
            c[j] = __builtin_amdgcn_wmma_f32_16x16x32_bf16(false, A.bf, false, B[j].bf,
                                                           (short)0, c[j], false, false);
    };

    const int steps = K >> 5;
    ldA(a0); ldB(b0);                    // stage 0
    for (int i = 1; i + 2 < steps; i += 2) {
        ldA(a1); ldB(b1);                // stage i   (loads overlap MMAs below)
        mma(a0, b0);                     // stage i-1
        ldA(a0); ldB(b0);                // stage i+1
        mma(a1, b1);                     // stage i
    }
    if (steps > 1) {
        ldA(a1); ldB(b1);                // last stage
        mma(a0, b0);
        mma(a1, b1);
    } else {
        mma(a0, b0);
    }

    const int rbase = (lane >> 4) * 8;     // C layout: lanes>=16 hold M=8..15
#pragma unroll
    for (int r = 0; r < 8; ++r) {
        int row = m0 + rbase + r;
        if (row < n) {
            float* yp = Y + (size_t)row * N + nb + col;
#pragma unroll
            for (int j = 0; j < 4; ++j) yp[16 * j] = c[j][r];
        }
    }
}

// ---------------- edge-phase kernels ----------------
// pass 1: alpha[e,h] = <q[dst,h,:], k[src,h,:] + ea[e]*We[h,:]> / 8 ; seg-max
__global__ void edge_alpha(const float* __restrict__ q, const float* __restrict__ k,
                           const float* __restrict__ ea, const float* __restrict__ We,
                           const int* __restrict__ ei, float* __restrict__ alpha,
                           float* __restrict__ m, int ne) {
    int t = blockIdx.x * blockDim.x + threadIdx.x;
    if (t >= ne * NHEAD) return;
    int e = t / NHEAD, h = t % NHEAD;
    int src = ei[e], dst = ei[ne + e];
    const float4* qd = (const float4*)(q  + (size_t)dst * FDIM + h * HIDC);
    const float4* ks = (const float4*)(k  + (size_t)src * FDIM + h * HIDC);
    const float4* we = (const float4*)(We + h * HIDC);
    float eav = ea[e];
    float acc = 0.0f;
#pragma unroll 4
    for (int c = 0; c < HIDC / 4; ++c) {
        float4 qv = qd[c], kv = ks[c], wv = we[c];
        acc += qv.x * (kv.x + eav * wv.x) + qv.y * (kv.y + eav * wv.y)
             + qv.z * (kv.z + eav * wv.z) + qv.w * (kv.w + eav * wv.w);
    }
    acc *= 0.125f;                      // 1/sqrt(64)
    alpha[t] = acc;
    atomicMaxF(&m[(size_t)dst * NHEAD + h], acc);
}

// pass 2: ex = exp(alpha - m[dst]); seg-sum
__global__ void edge_exp(const int* __restrict__ ei, const float* __restrict__ m,
                         float* __restrict__ alpha, float* __restrict__ s, int ne) {
    int t = blockIdx.x * blockDim.x + threadIdx.x;
    if (t >= ne * NHEAD) return;
    int e = t / NHEAD, h = t % NHEAD;
    int dst = ei[ne + e];
    float ex = __expf(alpha[t] - m[(size_t)dst * NHEAD + h]);
    alpha[t] = ex;
    atomicAdd(&s[(size_t)dst * NHEAD + h], ex);
}

// pass 3: out[dst,c] += (ex/(s+eps)) * (v[src,c] + ea*We[c]) for 4 channels/thread
__global__ void edge_aggr(const float* __restrict__ v, const float* __restrict__ ea,
                          const float* __restrict__ We, const int* __restrict__ ei,
                          const float* __restrict__ ex, const float* __restrict__ s,
                          float* __restrict__ out, int ne) {
    long t = (long)blockIdx.x * blockDim.x + threadIdx.x;
    if (t >= (long)ne * (FDIM / 4)) return;
    int e = (int)(t / (FDIM / 4)), c4 = (int)(t % (FDIM / 4));
    int c = c4 * 4;
    int h = c >> 6;
    int src = ei[e], dst = ei[ne + e];
    float a = ex[(size_t)e * NHEAD + h] / (s[(size_t)dst * NHEAD + h] + 1e-16f);
    float eav = ea[e];
    float4 vv = *(const float4*)(v + (size_t)src * FDIM + c);
    float4 wv = *(const float4*)(We + c);
    float* op = out + (size_t)dst * FDIM + c;
    atomicAdd(op + 0, a * (vv.x + eav * wv.x));
    atomicAdd(op + 1, a * (vv.y + eav * wv.y));
    atomicAdd(op + 2, a * (vv.z + eav * wv.z));
    atomicAdd(op + 3, a * (vv.w + eav * wv.w));
}

// o = 0.5 * (a + b)
__global__ void combine_mean(const float* __restrict__ a, const float* __restrict__ b,
                             float* __restrict__ o, long n) {
    long t = (long)blockIdx.x * blockDim.x + threadIdx.x;
    if (t < n) o[t] = 0.5f * (a[t] + b[t]);
}

// wave-per-node residual + layernorm over 64 channels (wave32, 2 ch/lane)
__global__ void residual_ln(const float* __restrict__ h, const float* __restrict__ res,
                            const float* __restrict__ g, const float* __restrict__ be,
                            float* __restrict__ o, int n, int useRes) {
    int node = (blockIdx.x * blockDim.x + threadIdx.x) >> 5;
    int lane = threadIdx.x & 31;
    if (node >= n) return;
    const float* hp = h + (size_t)node * HIDC;
    float v0 = hp[lane], v1 = hp[lane + 32];
    if (useRes) {
        const float* rp = res + (size_t)node * HIDC;
        v0 += rp[lane]; v1 += rp[lane + 32];
    }
    float ssum = v0 + v1;
#pragma unroll
    for (int off = 16; off; off >>= 1) ssum += __shfl_xor(ssum, off, 32);
    float mu = ssum * (1.0f / 64.0f);
    float d0 = v0 - mu, d1 = v1 - mu;
    float vs = d0 * d0 + d1 * d1;
#pragma unroll
    for (int off = 16; off; off >>= 1) vs += __shfl_xor(vs, off, 32);
    float inv = rsqrtf(vs * (1.0f / 64.0f) + 1e-5f);
    float* op = o + (size_t)node * HIDC;
    op[lane]      = d0 * inv * g[lane]      + be[lane];
    op[lane + 32] = d1 * inv * g[lane + 32] + be[lane + 32];
}

// global mean pool: accumulate sums and counts
__global__ void pool_kernel(const float* __restrict__ x, const int* __restrict__ batch,
                            float* __restrict__ pooled, float* __restrict__ cnt, int n) {
    long t = (long)blockIdx.x * blockDim.x + threadIdx.x;
    if (t >= (long)n * HIDC) return;
    int i = (int)(t >> 6), c = (int)(t & 63);
    int b = batch[i];
    atomicAdd(&pooled[(size_t)b * HIDC + c], x[t]);
    if (c == 0) atomicAdd(&cnt[b], 1.0f);
}

// relu(graph @ W1^T + b1) @ W2^T + b2 ;  one block per graph
__global__ void final_mlp(const float* __restrict__ pooled, const float* __restrict__ cnt,
                          const float* __restrict__ W1, const float* __restrict__ b1,
                          const float* __restrict__ W2, const float* __restrict__ b2,
                          float* __restrict__ out) {
    __shared__ float gv[HIDC];
    __shared__ float hv[HIDC];
    int gidx = blockIdx.x, t = threadIdx.x;
    float c = fmaxf(cnt[gidx], 1.0f);
    gv[t] = pooled[(size_t)gidx * HIDC + t] / c;
    __syncthreads();
    float acc = b1[t];
#pragma unroll 8
    for (int j = 0; j < HIDC; ++j) acc += gv[j] * W1[t * HIDC + j];
    hv[t] = fmaxf(acc, 0.0f);
    __syncthreads();
    if (t < 2) {
        float a2 = b2[t];
#pragma unroll 8
        for (int j = 0; j < HIDC; ++j) a2 += hv[j] * W2[t * HIDC + j];
        out[gidx * 2 + t] = a2;
    }
}

// ---------------- host side ----------------
struct ConvOff { size_t We, Wk, Wq, Ws, Wv, bk, bq, bs, bv; int si, di; };
struct BfOff   { size_t Wq, Wk, Wv, Ws; };

static inline unsigned nblk(long n) { return (unsigned)((n + 255) / 256); }

extern "C" void kernel_launch(void* const* d_in, const int* in_sizes, int n_in,
                              void* d_out, int out_size, void* d_ws, size_t ws_size,
                              hipStream_t stream) {
    const float* child_x = (const float*)d_in[0];
    const float* eaS  = (const float*)d_in[1];
    const float* eaR  = (const float*)d_in[2];
    const float* eaI  = (const float*)d_in[3];
    const float* P    = (const float*)d_in[4];   // params flattened (pytree order)
    const int*   eiS  = (const int*)d_in[5];     // [2,E] row-major: src then dst
    const int*   eiR  = (const int*)d_in[6];
    const int*   eiI  = (const int*)d_in[7];
    const int*   batch = (const int*)d_in[8];
    float* out = (float*)d_out;
    (void)in_sizes; (void)n_in; (void)out_size; (void)ws_size;

    // ---- param offsets (jax pytree flatten: dict keys sorted) ----
    size_t off = 0;
    size_t apEmb = off; off += (size_t)APEMB * EMBED;
    ConvOff co[3][3];   // [layer][0=intra, 1=rev, 2=sense]
    for (int i = 0; i < 3; ++i) {
        int ci = (i == 0) ? 1 : HIDC;
        int ai = (i == 0) ? EMBED : HIDC;
        int dims[3][2] = { {ci, ci}, {ai, ci}, {ci, ai} }; // {src,dst}
        for (int s3 = 0; s3 < 3; ++s3) {
            ConvOff& c = co[i][s3];
            c.si = dims[s3][0]; c.di = dims[s3][1];
            c.We = off; off += FDIM;
            c.Wk = off; off += (size_t)FDIM * c.si;
            c.Wq = off; off += (size_t)FDIM * c.di;
            c.Ws = off; off += (size_t)FDIM * c.di;
            c.Wv = off; off += (size_t)FDIM * c.si;
            c.bk = off; off += FDIM;
            c.bq = off; off += FDIM;
            c.bs = off; off += FDIM;
            c.bv = off; off += FDIM;
        }
    }
    size_t n_ap_g = off; off += HIDC;  size_t n_ap_b = off; off += HIDC;
    size_t n_ch_g = off; off += HIDC;  size_t n_ch_b = off; off += HIDC;
    size_t oW1 = off; off += HIDC * HIDC;  size_t ob1 = off; off += HIDC;
    size_t oW2 = off; off += 2 * HIDC;     size_t ob2 = off; off += 2;
    size_t pjW[3][2], pjB[3][2];           // [layer][0=ap, 1=child]
    for (int i = 0; i < 3; ++i) {
        pjW[i][0] = off; off += (size_t)HIDC * FDIM; pjB[i][0] = off; off += HIDC;
        pjW[i][1] = off; off += (size_t)HIDC * FDIM; pjB[i][1] = off; off += HIDC;
    }

    // ---- workspace carve ----
    char* base = (char*)d_ws; size_t wso = 0;
    auto alloc = [&](size_t bytes) -> char* {
        char* p = base + wso; wso = (wso + bytes + 255) & ~(size_t)255; return p;
    };
    float* P1 = (float*)alloc((size_t)NCHILD * FDIM * 4);
    float* P2 = (float*)alloc((size_t)NCHILD * FDIM * 4);
    float* P3 = (float*)alloc((size_t)NCHILD * FDIM * 4);
    float* P4 = (float*)alloc((size_t)NCHILD * FDIM * 4);
    float* A1 = (float*)alloc((size_t)NAP * FDIM * 4);
    float* A2 = (float*)alloc((size_t)NAP * FDIM * 4);
    float* A3 = (float*)alloc((size_t)NAP * FDIM * 4);
    float* xaA = (float*)alloc((size_t)NAP * HIDC * 4);
    float* xaB = (float*)alloc((size_t)NAP * HIDC * 4);
    float* xcA = (float*)alloc((size_t)NCHILD * HIDC * 4);
    float* xcB = (float*)alloc((size_t)NCHILD * HIDC * 4);
    float* exbuf  = (float*)alloc((size_t)NEDGE * NHEAD * 4);
    float* mbuf   = (float*)alloc((size_t)NCHILD * NHEAD * 4);
    float* sbuf   = (float*)alloc((size_t)NCHILD * NHEAD * 4);
    float* pooled = (float*)alloc((size_t)NG * HIDC * 4);
    float* cnt    = (float*)alloc((size_t)NG * 4);
    unsigned short* WBF  = (unsigned short*)alloc(4u << 20);          // bf16 weights
    unsigned short* XCBF = (unsigned short*)alloc((size_t)NCHILD * HIDC * 2);  // xc bf16
    unsigned short* XABF = (unsigned short*)alloc((size_t)NAP * HIDC * 2);     // xa bf16
    unsigned short* PCBF = (unsigned short*)alloc((size_t)NCHILD * FDIM * 2);  // proj-in child
    unsigned short* PABF = (unsigned short*)alloc((size_t)NAP * FDIM * 2);     // proj-in ap

    // ---- convert weights to transposed bf16 (WMMA B layout) ----
    size_t bfo = 0;
    auto cvtW = [&](size_t wOff, int outd, int ind) -> size_t {
        size_t r = bfo; bfo += (size_t)outd * ind;
        long n = (long)outd * ind;
        convert_wt<<<dim3(nblk(n)), dim3(256), 0, stream>>>(P + wOff, WBF + r, outd, ind);
        return r;
    };
    BfOff bco[3][3];
    for (int i = 0; i < 3; ++i)
        for (int s3 = 0; s3 < 3; ++s3) {
            const ConvOff& c = co[i][s3];
            BfOff& b = bco[i][s3];
            b.Wk = (c.si > 1) ? cvtW(c.Wk, FDIM, c.si) : 0;
            b.Wv = (c.si > 1) ? cvtW(c.Wv, FDIM, c.si) : 0;
            b.Wq = (c.di > 1) ? cvtW(c.Wq, FDIM, c.di) : 0;
            b.Ws = (c.di > 1) ? cvtW(c.Ws, FDIM, c.di) : 0;
        }
    size_t pjBf[3][2];
    for (int i = 0; i < 3; ++i) {
        pjBf[i][0] = cvtW(pjW[i][0], HIDC, FDIM);
        pjBf[i][1] = cvtW(pjW[i][1], HIDC, FDIM);
    }

    auto cvtX = [&](const float* X, unsigned short* Xb, long n) {
        convert_x<<<dim3(nblk(n)), dim3(256), 0, stream>>>(X, Xb, n);
    };
    auto gemmL = [&](const unsigned short* Xb, size_t wtOff, size_t bOff, float* Y,
                     int n, int K, int N) {
        dim3 g((unsigned)((n + 15) / 16), (unsigned)(N / 64));
        gemm_wmma<<<g, dim3(32), 0, stream>>>(Xb, WBF + wtOff, P + bOff, Y, n, K, N);
    };

    // run one TransformerConv; xs_bf/xd_bf are bf16 activations (unused if K==1)
    auto run_conv = [&](const float* xs, const unsigned short* xs_bf, int Ks,
                        const float* xd, const unsigned short* xd_bf, int Kd,
                        const ConvOff& c, const BfOff& b,
                        const int* ei, const float* ea, int nsrc, int ndst,
                        float* kbuf, float* vbuf, float* qbuf, float* obuf) {
        if (Ks == 1) {
            lin_k1<<<dim3(nblk((long)nsrc * FDIM)), dim3(256), 0, stream>>>(xs, P + c.Wk, P + c.bk, kbuf, nsrc);
            lin_k1<<<dim3(nblk((long)nsrc * FDIM)), dim3(256), 0, stream>>>(xs, P + c.Wv, P + c.bv, vbuf, nsrc);
        } else {
            gemmL(xs_bf, b.Wk, c.bk, kbuf, nsrc, Ks, FDIM);
            gemmL(xs_bf, b.Wv, c.bv, vbuf, nsrc, Ks, FDIM);
        }
        if (Kd == 1)
            lin_k1<<<dim3(nblk((long)ndst * FDIM)), dim3(256), 0, stream>>>(xd, P + c.Wq, P + c.bq, qbuf, ndst);
        else
            gemmL(xd_bf, b.Wq, c.bq, qbuf, ndst, Kd, FDIM);

        fill_f32<<<dim3(nblk((long)ndst * NHEAD)), dim3(256), 0, stream>>>(mbuf, -INFINITY, (long)ndst * NHEAD);
        hipMemsetAsync(sbuf, 0, (size_t)ndst * NHEAD * sizeof(float), stream);
        edge_alpha<<<dim3(nblk((long)NEDGE * NHEAD)), dim3(256), 0, stream>>>(qbuf, kbuf, ea, P + c.We, ei, exbuf, mbuf, NEDGE);
        edge_exp<<<dim3(nblk((long)NEDGE * NHEAD)), dim3(256), 0, stream>>>(ei, mbuf, exbuf, sbuf, NEDGE);

        if (Kd == 1)
            lin_k1<<<dim3(nblk((long)ndst * FDIM)), dim3(256), 0, stream>>>(xd, P + c.Ws, P + c.bs, obuf, ndst);
        else
            gemmL(xd_bf, b.Ws, c.bs, obuf, ndst, Kd, FDIM);
        edge_aggr<<<dim3(nblk((long)NEDGE * (FDIM / 4))), dim3(256), 0, stream>>>(vbuf, ea, P + c.We, ei, exbuf, sbuf, obuf, NEDGE);
    };

    // ---- initial features ----
    gather_ap_embed<<<dim3(nblk((long)NAP * EMBED)), dim3(256), 0, stream>>>(P + apEmb, xaA);

    const float* xc_cur = child_x; int Kc = 1;
    const float* xa_cur = xaA;     int Ka = EMBED;
    float* xc_nxt = xcA; float* xa_nxt = xaB;

    for (int L = 0; L < 3; ++L) {
        // pre-convert layer-input activations to bf16 (shared by all convs)
        cvtX(xa_cur, XABF, (long)NAP * Ka);
        if (Kc > 1) cvtX(xc_cur, XCBF, (long)NCHILD * Kc);

        // sense: child -> ap ; out -> A2
        run_conv(xc_cur, XCBF, Kc, xa_cur, XABF, Ka, co[L][2], bco[L][2], eiS, eaS,
                 NCHILD, NAP, P1, P2, A1, A2);
        // rev: ap -> child ; out -> P4
        run_conv(xa_cur, XABF, Ka, xc_cur, XCBF, Kc, co[L][1], bco[L][1], eiR, eaR,
                 NAP, NCHILD, A1, A3, P3, P4);
        // intra: child -> child ; out -> P1 (aliases kbuf; safe, skip runs after pass2)
        run_conv(xc_cur, XCBF, Kc, xc_cur, XCBF, Kc, co[L][0], bco[L][0], eiI, eaI,
                 NCHILD, NCHILD, P1, P2, P3, P1);
        // HeteroConv mean for child
        combine_mean<<<dim3(nblk((long)NCHILD * FDIM)), dim3(256), 0, stream>>>(P4, P1, P2, (long)NCHILD * FDIM);

        // projections (WMMA, K=384 -> N=64) + residual (layers 1,2) + layernorm
        cvtX(P2, PCBF, (long)NCHILD * FDIM);
        cvtX(A2, PABF, (long)NAP * FDIM);
        gemmL(PCBF, pjBf[L][1], pjB[L][1], P1, NCHILD, FDIM, HIDC);
        residual_ln<<<dim3(nblk((long)NCHILD * 32)), dim3(256), 0, stream>>>(
            P1, xc_cur, P + n_ch_g, P + n_ch_b, xc_nxt, NCHILD, (L > 0) ? 1 : 0);
        gemmL(PABF, pjBf[L][0], pjB[L][0], A1, NAP, FDIM, HIDC);
        residual_ln<<<dim3(nblk((long)NAP * 32)), dim3(256), 0, stream>>>(
            A1, xa_cur, P + n_ap_g, P + n_ap_b, xa_nxt, NAP, (L > 0) ? 1 : 0);

        xc_cur = xc_nxt; xa_cur = xa_nxt; Kc = HIDC; Ka = HIDC;
        xc_nxt = (xc_cur == xcA) ? xcB : xcA;
        xa_nxt = (xa_cur == xaA) ? xaB : xaA;
    }

    // ---- global mean pool + output MLP ----
    hipMemsetAsync(pooled, 0, (size_t)NG * HIDC * sizeof(float), stream);
    hipMemsetAsync(cnt,    0, (size_t)NG * sizeof(float), stream);
    pool_kernel<<<dim3(nblk((long)NCHILD * HIDC)), dim3(256), 0, stream>>>(xc_cur, batch, pooled, cnt, NCHILD);
    final_mlp<<<dim3(NG), dim3(HIDC), 0, stream>>>(pooled, cnt, P + oW1, P + ob1, P + oW2, P + ob2, out);
}
